// VectorQuantiser_67705864454614
// MI455X (gfx1250) — matrix operand order repaired and checked by reading, not compiled, not run on hardware
//
#include <hip/hip_runtime.h>
#include <hip/hip_bf16.h>

// Problem constants (from the reference):
//   x: [32,64,64,256] fp32  -> flat [131072, 256]
//   embeddings: [256, 1024] fp32
//   out = x + (gather(E, argmin_k dist) - x)   (fp32)
#define N_ROWS    131072
#define DIM       256
#define K_CODES   1024
#define TILE_ROWS 128          // rows per block (8 waves * 16 rows)
#define LDS_STRIDE 264         // 256 + 8 bf16 pad: keeps 16B alignment, spreads banks

typedef __attribute__((ext_vector_type(16))) __bf16 v16bf;
typedef __attribute__((ext_vector_type(8)))  __bf16 v8bf;
typedef __attribute__((ext_vector_type(8)))  float  v8f;

static __device__ __forceinline__ unsigned short f2bf(float f) {
  // round-to-nearest-even fp32 -> bf16
  unsigned u = __float_as_uint(f);
  u += 0x7FFFu + ((u >> 16) & 1u);
  return (unsigned short)(u >> 16);
}

// ---------------------------------------------------------------------------
// Prep: Et[k][d] = E[d][k] (fp32, for coalesced gather)
//       Ebt[k][d] = bf16(E[d][k]) (so B fragments are contiguous 32B loads)
//       enorm[k] = sum_d E[d][k]^2
// ---------------------------------------------------------------------------
__global__ __launch_bounds__(DIM) void vq_prep_kernel(
    const float* __restrict__ E,
    float* __restrict__ Et,
    unsigned short* __restrict__ Ebt,
    float* __restrict__ enorm) {
  const int k = blockIdx.x;
  const int d = threadIdx.x;
  const float v = E[(size_t)d * K_CODES + k];
  Et[(size_t)k * DIM + d]  = v;
  Ebt[(size_t)k * DIM + d] = f2bf(v);
  __shared__ float red[DIM];
  red[d] = v * v;
  __syncthreads();
  for (int s = DIM / 2; s > 0; s >>= 1) {
    if (d < s) red[d] += red[d + s];
    __syncthreads();
  }
  if (d == 0) enorm[k] = red[0];
}

// ---------------------------------------------------------------------------
// Main: per block, 128 rows. 8 waves, one 16-row slab per wave.
// Distances ranked by enorm[k] - 2*sim (||x||^2 constant per row -> dropped).
// sim via v_wmma_f32_16x16x32_bf16 accumulating over K=256 (8 fragments).
// ---------------------------------------------------------------------------
__global__ __launch_bounds__(256) void vq_main_kernel(
    const float* __restrict__ x,
    const float* __restrict__ Et,
    const unsigned short* __restrict__ Ebt,
    const float* __restrict__ enorm,
    float* __restrict__ out) {
  __shared__ __attribute__((aligned(16))) unsigned short ldsX[TILE_ROWS * LDS_STRIDE];
  __shared__ float enormLds[K_CODES];
  __shared__ int   ldsIdx[TILE_ROWS];

  const int tid    = threadIdx.x;
  const int wave   = tid >> 5;
  const int lane   = tid & 31;
  const int laneHi = lane >> 4;    // 0: lanes 0-15, 1: lanes 16-31
  const int laneLo = lane & 15;
  const size_t rowBase = (size_t)blockIdx.x * TILE_ROWS;

  // ---- Stage x tile -> LDS as bf16 (coalesced float4 reads) ----
  for (int it = 0; it < 32; ++it) {
    const int i4  = it * 1024 + tid * 4;     // flat element index in tile
    const int row = i4 >> 8;
    const int d   = i4 & 255;
    const float4 v = *(const float4*)(x + rowBase * DIM + (size_t)row * DIM + d);
    unsigned* p = (unsigned*)&ldsX[row * LDS_STRIDE + d];
    p[0] = (unsigned)f2bf(v.x) | ((unsigned)f2bf(v.y) << 16);
    p[1] = (unsigned)f2bf(v.z) | ((unsigned)f2bf(v.w) << 16);
  }
  for (int i = tid; i < K_CODES; i += 256) enormLds[i] = enorm[i];
  __syncthreads();

  // ---- Load A fragments (16 rows x K=256) into registers ----
  // A 16x32 layout (matching B): lane L holds row M=L%16;
  // K = (L/16)*16 + (0..15) sequentially -> two contiguous 16B LDS reads.
  v16bf a[8];
#pragma unroll
  for (int t = 0; t < 8; ++t) {
    const unsigned short* base =
        &ldsX[(wave * 16 + laneLo) * LDS_STRIDE + t * 32 + laneHi * 16];
    const v8bf lo = *(const v8bf*)(base);        // K = koff+0..7
    const v8bf hi = *(const v8bf*)(base + 8);    // K = koff+8..15
    v16bf av;
#pragma unroll
    for (int i = 0; i < 8; ++i) { av[i] = lo[i]; av[i + 8] = hi[i]; }
    a[t] = av;
  }

  float bestD[8];
  int   bestI[8];
#pragma unroll
  for (int r = 0; r < 8; ++r) { bestD[r] = 3.4e38f; bestI[r] = 0; }

  // ---- Sweep the 1024 codes in 16-wide tiles ----
  for (int kt = 0; kt < K_CODES / 16; ++kt) {
    const int col = kt * 16 + laneLo;
    // B 32x16 layout used here (same k order as A): lane L holds column N=L%16,
    // K = (L/16)*16 + (0..15) sequentially -> one contiguous v16bf load.
    const unsigned short* bbase = Ebt + (size_t)col * DIM + laneHi * 16;
    if (kt < K_CODES / 16 - 1)
      __builtin_prefetch(bbase + 16 * DIM, 0, 0);   // next tile's column

    v8f c = {};
#pragma unroll
    for (int t = 0; t < 8; ++t) {
      const v16bf b = *(const v16bf*)(bbase + t * 32);
      c = __builtin_amdgcn_wmma_f32_16x16x32_bf16(
              false, a[t], false, b, (short)0, c, false, false);
    }
    // C layout: lane L, VGPR r holds sim[row = r + 8*(L/16)][col = L%16]
    const float en = enormLds[col];
#pragma unroll
    for (int r = 0; r < 8; ++r) {
      const float dval = en - 2.0f * c[r];
      if (dval < bestD[r]) { bestD[r] = dval; bestI[r] = col; }
    }
  }

  // ---- Cross-lane argmin within each 16-lane half (wave32) ----
#pragma unroll
  for (int off = 1; off < 16; off <<= 1) {
#pragma unroll
    for (int r = 0; r < 8; ++r) {
      const float od = __shfl_xor(bestD[r], off, 32);
      const int   oi = __shfl_xor(bestI[r], off, 32);
      if (od < bestD[r] || (od == bestD[r] && oi < bestI[r])) {
        bestD[r] = od; bestI[r] = oi;
      }
    }
  }
  if (laneLo == 0) {
#pragma unroll
    for (int r = 0; r < 8; ++r)
      ldsIdx[wave * 16 + laneHi * 8 + r] = bestI[r];
  }
  __syncthreads();

  // ---- Gather fp32 codebook rows + straight-through estimator ----
  for (int r2 = 0; r2 < 16; ++r2) {
    const int ksel = ldsIdx[wave * 16 + r2];
    const float* q = Et + (size_t)ksel * DIM;       // contiguous, L2-resident
    const size_t row = rowBase + (size_t)(wave * 16 + r2);
#pragma unroll
    for (int j = 0; j < 8; ++j) {
      const int d = j * 32 + lane;                  // coalesced
      const float xv = x[row * DIM + d];
      out[row * DIM + d] = xv + (q[d] - xv);        // matches reference fp math
    }
  }
}

// ---------------------------------------------------------------------------
extern "C" void kernel_launch(void* const* d_in, const int* in_sizes, int n_in,
                              void* d_out, int out_size, void* d_ws, size_t ws_size,
                              hipStream_t stream) {
  (void)in_sizes; (void)n_in; (void)out_size; (void)ws_size;
  const float* x = (const float*)d_in[0];          // [131072, 256]
  const float* E = (const float*)d_in[1];          // [256, 1024]
  float* out = (float*)d_out;

  // Workspace layout: Et (1 MB) | Ebt (512 KB) | enorm (4 KB)
  char* ws = (char*)d_ws;
  float*          Et    = (float*)ws;
  unsigned short* Ebt   = (unsigned short*)(ws + (size_t)K_CODES * DIM * sizeof(float));
  float*          enorm = (float*)(ws + (size_t)K_CODES * DIM * (sizeof(float) + sizeof(unsigned short)));

  vq_prep_kernel<<<K_CODES, DIM, 0, stream>>>(E, Et, Ebt, enorm);
  vq_main_kernel<<<N_ROWS / TILE_ROWS, 256, 0, stream>>>(x, Et, Ebt, enorm, out);
}
